// EF_expansion_18107582120608
// MI455X (gfx1250) — matrix operand order repaired and checked by reading, not compiled, not run on hardware
//
#include <hip/hip_runtime.h>
#include <hip/hip_bf16.h>
#include <math.h>

// Shapes (fixed by the reference)
#define Bn 16
#define Cc 128
#define Nn 2048
#define Kk 4
#define Oo 64
#define Rr 2

typedef __attribute__((ext_vector_type(16))) __bf16 v16bf;
typedef __attribute__((ext_vector_type(8)))  __bf16 v8bf;
typedef __attribute__((ext_vector_type(8)))  float  v8f;

#define WMMA_BF16(A_, B_, C_) \
  __builtin_amdgcn_wmma_f32_16x16x32_bf16(false, (A_), false, (B_), (short)0, (C_), false, false)

// Single-wave LDS store->load fence: DS ops of one wave are in-order (DScnt),
// so s_wait_dscnt 0 alone gives visibility -- avoids the full
// s_wait_loadcnt_dscnt 0 a workgroup barrier would emit (which would force
// our global prefetches to land before the scan).
__device__ __forceinline__ void wave_lds_fence() {
  asm volatile("s_wait_dscnt 0x0" ::: "memory");
}
// Compiler-only ordering barrier (hardware keeps DS in order per wave).
__device__ __forceinline__ void compiler_fence() {
  asm volatile("" ::: "memory");
}

// ---------------------------------------------------------------------------
// Kernel 1: x [B][C][N] f32 -> xbT [B][N][C] bf16, LDS-tiled 64x64 transpose
// (coalesced f32 reads, coalesced contiguous bf16 writes)
// ---------------------------------------------------------------------------
__global__ __launch_bounds__(256) void cvt_kernel(const float* __restrict__ x,
                                                  __bf16* __restrict__ xbT) {
  __shared__ float tile[64 * 65];
  int blk = blockIdx.x;                    // B * (N/64) * (C/64) = 1024
  int b   = blk >> 6;
  int rem = blk & 63;
  int n0 = (rem & 31) << 6;                // 32 n-tiles
  int c0 = (rem >> 5) << 6;                // 2 c-tiles
  int t = threadIdx.x;
  #pragma unroll
  for (int i = 0; i < 16; ++i) {           // read: consecutive lanes -> consec n
    int lin = t + i * 256;
    int c = lin >> 6, n = lin & 63;
    tile[c * 65 + n] = x[((size_t)b * Cc + c0 + c) * Nn + n0 + n];
  }
  __syncthreads();                         // multi-wave block: real barrier
  #pragma unroll
  for (int i = 0; i < 16; ++i) {           // write: consecutive lanes -> consec c
    int lin = t + i * 256;
    int n = lin >> 6, c = lin & 63;
    xbT[((size_t)b * Nn + n0 + n) * Cc + c0 + c] = (__bf16)tile[c * 65 + n];
  }
}

// ---------------------------------------------------------------------------
// Kernel 2: xx[p] = sum_c bf16(x)^2  (bf16-consistent with the WMMA Gram so
// pd(i,i) == 0 exactly and self lands in the top-4, like the reference)
// ---------------------------------------------------------------------------
__global__ __launch_bounds__(256) void xx_kernel(const __bf16* __restrict__ xbT,
                                                 float* __restrict__ xx) {
  int p = blockIdx.x * 256 + threadIdx.x;  // over B*N = 32768
  const __bf16* v = xbT + (size_t)p * Cc;
  float s = 0.0f;
  for (int c = 0; c < Cc; ++c) { float f = (float)v[c]; s += f * f; }
  xx[p] = s;
}

// ---------------------------------------------------------------------------
// Kernel 3: pre-swizzle W[M][Kdim] (f32 row-major) into the CDNA5 16x32 bf16
// A-fragment in this file's contiguous k order (B loads match): lane l (h=l>=16), element e:
//   K = h*16 + e,  M = l%16.   dst[tile][lane][16], 1KB/tile.
// ---------------------------------------------------------------------------
__global__ __launch_bounds__(256) void swizzle_w(const float* __restrict__ W,
                                                 __bf16* __restrict__ dst,
                                                 int Mtiles, int Ktiles, int Kdim) {
  int t = blockIdx.x * 256 + threadIdx.x;
  int total = Mtiles * Ktiles * 32;
  if (t >= total) return;
  int lane = t & 31;
  int tile = t >> 5;
  int kt = tile % Ktiles;
  int mt = tile / Ktiles;
  int h = (lane >> 4) & 1;
  int m = mt * 16 + (lane & 15);
  __bf16* o = dst + (size_t)tile * 512 + lane * 16;
  #pragma unroll
  for (int e = 0; e < 16; ++e) {
    int K = kt * 32 + h * 16 + e;
    o[e] = (__bf16)W[(size_t)m * Kdim + K];
  }
}

// ---------------------------------------------------------------------------
// Kernel 4: fused Gram + top-4 KNN.  One wave32 per (batch, 32-row tile):
//  - 8 A fragments resident (32 rows x 128 ch)
//  - unroll-by-2 ping-pong B fragments: prefetch of tile k writes directly
//    into the buffer consumed by tile k+1 (no register copies at loop tail);
//    the prefetch now overlaps WMMA + LDS spill + scan (dscnt-only fence)
//  - 8 x v_wmma_f32_16x16x32_bf16 per 16-column tile
//  - xx[n] folded into the pd tile at store time (1 fma/elem)
//  - every lane owns one row: guarded branchless top-4 insert
// ---------------------------------------------------------------------------
__global__ __launch_bounds__(32) void knn_kernel(const __bf16* __restrict__ xbT,
                                                 const float* __restrict__ xx,
                                                 int* __restrict__ idx) {
  __shared__ float pdb[32 * 17];
  int l = threadIdx.x;
  int j = l & 15;
  int h = l >> 4;
  int blk = blockIdx.x;
  int b = blk >> 6;                        // N/32 = 64 row-tiles per batch
  int rowbase = (blk & 63) << 5;
  const __bf16* xb = xbT + (size_t)b * Nn * Cc;
  const float* xxb = xx + (size_t)b * Nn;

  // A fragments: two 16-row tiles, 4 ktiles each
  v16bf A[8];
  #pragma unroll
  for (int rt = 0; rt < 2; ++rt) {
    const __bf16* ap = xb + (size_t)(rowbase + rt * 16 + j) * Cc;
    #pragma unroll
    for (int kt = 0; kt < 4; ++kt) {
      v8bf lo = *(const v8bf*)(ap + kt * 32 + h * 16);
      v8bf hi = *(const v8bf*)(ap + kt * 32 + h * 16 + 8);
      A[rt * 4 + kt] =
          __builtin_shufflevector(lo, hi, 0,1,2,3,4,5,6,7,8,9,10,11,12,13,14,15);
    }
  }
  float xxm = xxb[rowbase + l];            // this lane's row norm

  float v0 = -INFINITY, v1 = -INFINITY, v2 = -INFINITY, v3 = -INFINITY;
  int   i0 = 0, i1 = 0, i2 = 0, i3 = 0;

  // prime the pipeline: tile 0 B fragments + column norms
  v16bf B0, B1, B2, B3, C0, C1, C2, C3;
  float xxn, xxn2;
  {
    const __bf16* bp = xb + (size_t)j * Cc + h * 16;
    B0 = *(const v16bf*)(bp);
    B1 = *(const v16bf*)(bp + 32);
    B2 = *(const v16bf*)(bp + 64);
    B3 = *(const v16bf*)(bp + 96);
    xxn = xxb[j];
  }

  // One column-tile step: consume (CUR*, XN), prefetch tile NT+1 into (NXT*)
#define KNN_TILE(NT, CUR0, CUR1, CUR2, CUR3, XN, NXT0, NXT1, NXT2, NXT3, NXN)  \
  {                                                                            \
    v8f acc0 = {}, acc1 = {};                                                  \
    acc0 = WMMA_BF16(A[0], CUR0, acc0);                                        \
    acc1 = WMMA_BF16(A[4], CUR0, acc1);                                        \
    acc0 = WMMA_BF16(A[1], CUR1, acc0);                                        \
    acc1 = WMMA_BF16(A[5], CUR1, acc1);                                        \
    acc0 = WMMA_BF16(A[2], CUR2, acc0);                                        \
    acc1 = WMMA_BF16(A[6], CUR2, acc1);                                        \
    acc0 = WMMA_BF16(A[3], CUR3, acc0);                                        \
    acc1 = WMMA_BF16(A[7], CUR3, acc1);                                        \
    int nn = ((NT) + 1) & 127;                                                 \
    const __bf16* bp = xb + (size_t)(nn * 16 + j) * Cc + h * 16;               \
    NXT0 = *(const v16bf*)(bp);                                                \
    NXT1 = *(const v16bf*)(bp + 32);                                           \
    NXT2 = *(const v16bf*)(bp + 64);                                           \
    NXT3 = *(const v16bf*)(bp + 96);                                           \
    NXN  = xxb[nn * 16 + j];                                                   \
    _Pragma("unroll")                                                          \
    for (int v = 0; v < 8; ++v)                                                \
      pdb[(v + 8 * h) * 17 + j] = 2.0f * acc0[v] - XN;                         \
    _Pragma("unroll")                                                          \
    for (int v = 0; v < 8; ++v)                                                \
      pdb[(16 + v + 8 * h) * 17 + j] = 2.0f * acc1[v] - XN;                    \
    wave_lds_fence();     /* dscnt-only: prefetch loads keep flying */         \
    int nbase = (NT) << 4;                                                     \
    _Pragma("unroll")                                                          \
    for (int jj = 0; jj < 16; ++jj) {                                          \
      float val = pdb[l * 17 + jj] - xxm;                                      \
      if (val > v3) {                                                          \
        int id = nbase + jj;                                                   \
        bool g0 = val > v0, g1 = val > v1, g2 = val > v2;                      \
        v3 = g2 ? v2 : val;              i3 = g2 ? i2 : id;                    \
        v2 = g2 ? (g1 ? v1 : val) : v2;  i2 = g2 ? (g1 ? i1 : id) : i2;        \
        v1 = g1 ? (g0 ? v0 : val) : v1;  i1 = g1 ? (g0 ? i0 : id) : i1;        \
        v0 = g0 ? val : v0;              i0 = g0 ? id : i0;                    \
      }                                                                        \
    }                                                                          \
    compiler_fence();     /* keep next tile's DS stores after these reads */   \
  }

  for (int nt = 0; nt < 128; nt += 2) {
    KNN_TILE(nt,     B0, B1, B2, B3, xxn,  C0, C1, C2, C3, xxn2)
    KNN_TILE(nt + 1, C0, C1, C2, C3, xxn2, B0, B1, B2, B3, xxn)
  }
#undef KNN_TILE

  int4 r; r.x = i0; r.y = i1; r.z = i2; r.w = i3;
  *(int4*)(idx + ((size_t)b * Nn + rowbase + l) * 4) = r;
}

// ---------------------------------------------------------------------------
// Kernel 5: fused MLP chain.  One wave32 per (batch, 4 points) handling the
// 16 columns (kk=0..3) x (ni=0..3), col j = kk*4 + ni.
//   GEMM1 (64x256)  -> relu -> LDS e1T[16][68]
//   GEMM2 (128x320) -> relu -> LDS e2T[16][132]   (K: 0..63 e1, 64..319 ief)
//   GEMM3 (64x64) x2 (R-split halves) + b3 -> max over kk via shfl_xor -> out
// ---------------------------------------------------------------------------
__global__ __launch_bounds__(32) void mlp_kernel(
    const __bf16* __restrict__ xbT, const int* __restrict__ idx,
    const __bf16* __restrict__ w1s, const __bf16* __restrict__ w2s,
    const __bf16* __restrict__ w3s,
    const float* __restrict__ b1, const float* __restrict__ b2,
    const float* __restrict__ b3, float* __restrict__ out) {
  __shared__ float e1T[16 * 68];    // [col][chan 0..63], stride 68 (16B aligned)
  __shared__ float e2T[16 * 132];   // [col][chan 0..127], stride 132

  int l = threadIdx.x;
  int j = l & 15;
  int h = l >> 4;
  int blk = blockIdx.x;
  int b = blk >> 9;                               // N/4 = 512
  int nbase4 = (blk & 511) << 2;
  int ni = j & 3;
  int kk = j >> 2;
  int ncen = nbase4 + ni;
  int nng  = idx[((size_t)b * Nn + ncen) * 4 + kk];
  const __bf16* xb = xbT + (size_t)b * Nn * Cc;
  const __bf16* pc = xb + (size_t)ncen * Cc;      // center point features
  const __bf16* pn = xb + (size_t)nng * Cc;       // neighbor point features

  // -------- GEMM1: e1 = w1 * ief + b1  (ief chan: 0..127 center, 128..255 nb)
  v16bf Bf[8];
  #pragma unroll
  for (int kt = 0; kt < 8; ++kt) {
    const __bf16* src = (kt < 4) ? pc : pn;
    Bf[kt] = *(const v16bf*)(src + (kt & 3) * 32 + h * 16);
  }
  #pragma unroll
  for (int mt = 0; mt < 4; ++mt) {
    v8f acc;
    #pragma unroll
    for (int v = 0; v < 8; ++v) acc[v] = b1[mt * 16 + v + 8 * h];
    #pragma unroll
    for (int kt = 0; kt < 8; ++kt) {
      v16bf Af = *(const v16bf*)(w1s + (size_t)(mt * 8 + kt) * 512 + l * 16);
      acc = WMMA_BF16(Af, Bf[kt], acc);
    }
    #pragma unroll
    for (int v = 0; v < 8; ++v)
      e1T[j * 68 + mt * 16 + v + 8 * h] = fmaxf(acc[v], 0.0f);  // relu(e1)
  }
  wave_lds_fence();

  // -------- GEMM2: e2 = relu(w2 * relu([e1; ief]) + b2)
  v16bf Bf2[10];
  #pragma unroll
  for (int kt = 0; kt < 2; ++kt) {                // K 0..63 from relu(e1) in LDS
    union { v16bf v; __bf16 e[16]; } u;
    #pragma unroll
    for (int e = 0; e < 16; ++e)
      u.e[e] = (__bf16)e1T[j * 68 + kt * 32 + h * 16 + e];
    Bf2[kt] = u.v;
  }
  #pragma unroll
  for (int kt = 2; kt < 10; ++kt) {               // K 64..319 from relu(ief)
    int cg = kt * 32 + h * 16 - 64;               // 0..255
    const __bf16* src = (cg < 128) ? pc : pn;
    union { v16bf v; unsigned short s[16]; } u;
    u.v = *(const v16bf*)(src + (cg & 127));
    #pragma unroll
    for (int e = 0; e < 16; ++e)
      if (u.s[e] & 0x8000) u.s[e] = 0;            // bf16 relu (sign-bit clear)
    Bf2[kt] = u.v;
  }
  #pragma unroll
  for (int mt = 0; mt < 8; ++mt) {
    v8f acc;
    #pragma unroll
    for (int v = 0; v < 8; ++v) acc[v] = b2[mt * 16 + v + 8 * h];
    #pragma unroll
    for (int kt = 0; kt < 10; ++kt) {
      v16bf Af = *(const v16bf*)(w2s + (size_t)(mt * 10 + kt) * 512 + l * 16);
      acc = WMMA_BF16(Af, Bf2[kt], acc);
    }
    #pragma unroll
    for (int v = 0; v < 8; ++v)
      e2T[j * 132 + mt * 16 + v + 8 * h] = fmaxf(acc[v], 0.0f); // relu(e2)
  }
  wave_lds_fence();

  // -------- GEMM3 per R-half + bias b3, then max over kk, write out
  #pragma unroll
  for (int r = 0; r < 2; ++r) {
    v16bf B3[2];
    #pragma unroll
    for (int kt = 0; kt < 2; ++kt) {
      union { v16bf v; __bf16 e[16]; } u;
      #pragma unroll
      for (int e = 0; e < 16; ++e)
        u.e[e] = (__bf16)e2T[j * 132 + r * 64 + kt * 32 + h * 16 + e];
      B3[kt] = u.v;
    }
    #pragma unroll
    for (int mt = 0; mt < 4; ++mt) {
      v8f acc;
      #pragma unroll
      for (int v = 0; v < 8; ++v) acc[v] = b3[mt * 16 + v + 8 * h];
      #pragma unroll
      for (int kt = 0; kt < 2; ++kt) {
        v16bf Af = *(const v16bf*)(w3s + (size_t)(mt * 2 + kt) * 512 + l * 16);
        acc = WMMA_BF16(Af, B3[kt], acc);
      }
      #pragma unroll
      for (int v = 0; v < 8; ++v) {
        float m = acc[v];
        m = fmaxf(m, __shfl_xor(m, 4));           // reduce over kk bit 0
        m = fmaxf(m, __shfl_xor(m, 8));           // reduce over kk bit 1
        if (kk == 0) {                            // one writer per (ni, row)
          int o = mt * 16 + v + 8 * h;
          out[((size_t)b * Oo + o) * (Nn * Rr) + 2 * (nbase4 + ni) + r] = m;
        }
      }
    }
  }
}

// ---------------------------------------------------------------------------
// Host launcher
// ---------------------------------------------------------------------------
extern "C" void kernel_launch(void* const* d_in, const int* in_sizes, int n_in,
                              void* d_out, int out_size, void* d_ws, size_t ws_size,
                              hipStream_t stream) {
  const float* x  = (const float*)d_in[0];
  const float* w1 = (const float*)d_in[1];
  const float* b1 = (const float*)d_in[2];
  const float* w2 = (const float*)d_in[3];
  const float* b2 = (const float*)d_in[4];
  const float* w3 = (const float*)d_in[5];
  const float* b3 = (const float*)d_in[6];
  float* out = (float*)d_out;

  // Workspace carve-out (~9.2 MB total, all 256B-aligned)
  char* ws = (char*)d_ws;
  __bf16* xbT = (__bf16*)(ws);                    // 8,388,608 B  [B][N][C] bf16
  float*  xx  = (float*)(ws + 8388608);           //   131,072 B  [B][N] f32
  int*    idx = (int*)(ws + 8519680);             //   524,288 B  [B][N][4] int
  __bf16* w1s = (__bf16*)(ws + 9043968);          //    32,768 B  swizzled w1
  __bf16* w2s = (__bf16*)(ws + 9076736);          //    81,920 B  swizzled w2
  __bf16* w3s = (__bf16*)(ws + 9158656);          //     8,192 B  swizzled w3

  cvt_kernel<<<Bn * (Nn / 64) * (Cc / 64), 256, 0, stream>>>(x, xbT);
  xx_kernel<<<(Bn * Nn) / 256, 256, 0, stream>>>(xbT, xx);
  swizzle_w<<<4, 256, 0, stream>>>(w1, w1s, 4, 8, 2 * Cc);          // 64x256
  swizzle_w<<<10, 256, 0, stream>>>(w2, w2s, 8, 10, 2 * Cc + Oo);   // 128x320
  swizzle_w<<<1, 256, 0, stream>>>(w3, w3s, 4, 2, Oo);              // 64x64
  knn_kernel<<<Bn * (Nn / 32), 32, 0, stream>>>(xbT, xx, idx);
  mlp_kernel<<<Bn * (Nn / 4), 32, 0, stream>>>(xbT, idx, w1s, w2s, w3s,
                                               b1, b2, b3, out);
}
